// GloVeRGAT_39273180955004
// MI455X (gfx1250) — matrix operand order repaired and checked by reading, not compiled, not run on hardware
//
#include <hip/hip_runtime.h>
#include <hip/hip_bf16.h>

// ---------------------------------------------------------------------------
// Sizes (fixed by the reference)
// ---------------------------------------------------------------------------
#define NN    4096
#define EE    65536
#define RR    8
#define EMB   300
#define HH    100
#define HEADS 8
#define HC    800     // H*HEADS
#define LL    8
#define G4H   400     // 4*H

typedef __attribute__((ext_vector_type(16))) _Float16 v16h;
typedef __attribute__((ext_vector_type(8)))  _Float16 v8h;
typedef __attribute__((ext_vector_type(8)))  float    v8f;

// ---------------------------------------------------------------------------
// WMMA GEMM:  C[M,Ntot] = act( A[M,K] * B^T + bias ), B addressed as
// B[col*sN + k*sK]  (covers both weight-major [N,K] and K-major [K,N]).
// Optional row gather on A (rowidx). Optional batch dim (blockIdx.z).
// Requirements (all call sites satisfy): M % 64 == 0, lda % 4 == 0,
// sN % 4 == 0 when sK == 1.
// 64x64 C tile / 256 threads; f32->f16 staged through double-buffered LDS;
// branch-free (clamp+select) guards so loads pipeline.
// ---------------------------------------------------------------------------
__global__ __launch_bounds__(256) void wmma_gemm_nt(
    const float* __restrict__ A, int lda, const int* __restrict__ rowidx,
    const float* __restrict__ B, long sN, long sK, long bStrideB,
    float* __restrict__ C, int ldc, long bStrideC,
    const float* __restrict__ bias, int M, int Ntot, int K, int doRelu)
{
  __shared__ __attribute__((aligned(32))) _Float16 As[2][64][32];
  __shared__ __attribute__((aligned(32))) _Float16 Bs[2][64][32];

  const int tid = threadIdx.x;
  const int bm  = blockIdx.y * 64;
  const int bn  = blockIdx.x * 64;
  const float* Bb = B + (long)blockIdx.z * bStrideB;
  float*       Cb = C + (long)blockIdx.z * bStrideC;

  // staging assignment: 4 threads per row/col, 8 contiguous k each
  const int ar = tid >> 2;
  const int ak = (tid & 3) * 8;
  const long arow  = rowidx ? (long)rowidx[bm + ar] : (long)(bm + ar);
  const float* Arow = A + arow * (long)lda;

  const int  bc    = tid >> 2;
  const int  bk    = (tid & 3) * 8;
  const int  colr  = bn + bc;
  const bool colOK = colr < Ntot;
  const int  colc  = colOK ? colr : (Ntot - 1);
  const float* Bcol = Bb + (long)colc * sN;

  // fragment geometry (ISA 7.12.2, wave32)
  const int wave = tid >> 5;
  const int lane = tid & 31;
  const int m0   = (wave & 3) * 16;
  const int n0   = (wave >> 2) * 32;
  const int rsel = lane & 15;
  const int hi   = lane >> 4;     // 0 | 1
  const int kbA  = hi * 16;
  const int kbB  = hi * 16;

  float fa[8], fb[8];

  auto loadTile = [&](int k0) {
    if (k0 + ak + 8 <= K) {               // fast: two 16B vector loads
      const float4* p = (const float4*)(Arow + k0 + ak);
      const float4 a0 = p[0], a1 = p[1];
      fa[0] = a0.x; fa[1] = a0.y; fa[2] = a0.z; fa[3] = a0.w;
      fa[4] = a1.x; fa[5] = a1.y; fa[6] = a1.z; fa[7] = a1.w;
    } else {                              // K tail: clamp + select (no branch)
#pragma unroll
      for (int i = 0; i < 8; i++) {
        const int k = k0 + ak + i;
        const int kc = (k < K) ? k : (K - 1);
        const float v = Arow[kc];
        fa[i] = (k < K) ? v : 0.f;
      }
    }
    if (sK == 1 && k0 + bk + 8 <= K) {    // fast: contiguous weight rows
      const float4* p = (const float4*)(Bcol + k0 + bk);
      const float4 b0 = p[0], b1 = p[1];
      fb[0] = b0.x; fb[1] = b0.y; fb[2] = b0.z; fb[3] = b0.w;
      fb[4] = b1.x; fb[5] = b1.y; fb[6] = b1.z; fb[7] = b1.w;
#pragma unroll
      for (int i = 0; i < 8; i++) fb[i] = colOK ? fb[i] : 0.f;
    } else {                              // strided (K-major) or K tail
#pragma unroll
      for (int i = 0; i < 8; i++) {
        const int k = k0 + bk + i;
        const int kc = (k < K) ? k : (K - 1);
        const float v = Bcol[(long)kc * sK];
        fb[i] = (colOK && k < K) ? v : 0.f;
      }
    }
  };

  auto stage = [&](int buf) {
    v8h ha, hb;
#pragma unroll
    for (int i = 0; i < 8; i++) { ha[i] = (_Float16)fa[i]; hb[i] = (_Float16)fb[i]; }
    *(v8h*)&As[buf][ar][ak] = ha;        // one ds_store_b128
    *(v8h*)&Bs[buf][bc][bk] = hb;        // one ds_store_b128
  };

  v8f acc0 = {};
  v8f acc1 = {};

  const int ktiles = (K + 31) / 32;
  loadTile(0);
  stage(0);
  __syncthreads();

  for (int kt = 0; kt < ktiles; kt++) {
    const int buf = kt & 1;
    const int nxt = kt + 1;
    if (nxt < ktiles) loadTile(nxt * 32);   // global loads overlap WMMA

    union { v16h v; v8h h[2]; } ua;
    ua.h[0] = *(const v8h*)&As[buf][m0 + rsel][kbA];
    ua.h[1] = *(const v8h*)&As[buf][m0 + rsel][kbA + 8];
    v16h bf0 = *(const v16h*)&Bs[buf][n0 + rsel][kbB];
    v16h bf1 = *(const v16h*)&Bs[buf][n0 + 16 + rsel][kbB];

    acc0 = __builtin_amdgcn_wmma_f32_16x16x32_f16(false, ua.v, false, bf0,
                                                  (short)0, acc0, false, false);
    acc1 = __builtin_amdgcn_wmma_f32_16x16x32_f16(false, ua.v, false, bf1,
                                                  (short)0, acc1, false, false);

    if (nxt < ktiles) stage(nxt & 1);       // write the other buffer
    __syncthreads();
  }

#pragma unroll
  for (int r = 0; r < 8; r++) {
    const int m = bm + m0 + r + hi * 8;     // M % 64 == 0 -> always valid
    const int n = bn + n0 + rsel;
    if (n < Ntot) {
      float v = acc0[r] + (bias ? bias[n] : 0.f);
      Cb[(long)m * ldc + n] = doRelu ? fmaxf(v, 0.f) : v;
    }
    const int n2 = n + 16;
    if (n2 < Ntot) {
      float v = acc1[r] + (bias ? bias[n2] : 0.f);
      Cb[(long)m * ldc + n2] = doRelu ? fmaxf(v, 0.f) : v;
    }
  }
}

// ---------------------------------------------------------------------------
// LSTM weight transpose: [L][400][100] -> [L][100][400] (coalesced matvec)
// ---------------------------------------------------------------------------
__global__ void transpose_lstm(const float* __restrict__ src,
                               float* __restrict__ dst, int Lc)
{
  const int idx = blockIdx.x * blockDim.x + threadIdx.x;
  const int tot = Lc * G4H * HH;
  if (idx >= tot) return;
  const int l  = idx / (G4H * HH);
  const int rem = idx % (G4H * HH);
  const int j  = rem / HH;
  const int k2 = rem % HH;
  dst[l * (G4H * HH) + k2 * G4H + j] = src[idx];
}

// ---------------------------------------------------------------------------
// Sequential stacked-LSTM scan. One workgroup; h/c state in LDS; layer-0
// input projection (xproj = x_t @ w_ih0^T) precomputed by WMMA GEMM.
// hc global buffer carries (h,c) between the 4 lstm_run phases.
// ---------------------------------------------------------------------------
__global__ __launch_bounds__(512) void lstm_scan(
    const float* __restrict__ xproj, const float* __restrict__ wTih,
    const float* __restrict__ wThh,  const float* __restrict__ b_ih,
    const float* __restrict__ b_hh,  float* __restrict__ out,
    float* __restrict__ hc, int T, int initState)
{
  __shared__ float hs[LL][HH];
  __shared__ float cs[LL][HH];
  __shared__ float gates[G4H];
  __shared__ float bsum[LL * G4H];

  const int tid = threadIdx.x;
  for (int i = tid; i < LL * G4H; i += 512) bsum[i] = b_ih[i] + b_hh[i];
  for (int i = tid; i < LL * HH; i += 512) {
    const int l = i / HH, k2 = i % HH;
    hs[l][k2] = initState ? 0.f : hc[i];
    cs[l][k2] = initState ? 0.f : hc[LL * HH + i];
  }
  __syncthreads();

  for (int t = 0; t < T; t++) {
    for (int l = 0; l < LL; l++) {
      if (tid < G4H) {
        float s = bsum[l * G4H + tid];
        if (l == 0) {
          s += xproj[(long)t * G4H + tid];
        } else {
          const float* wi = wTih + (long)(l - 1) * (G4H * HH);
#pragma unroll 4
          for (int k2 = 0; k2 < HH; k2++)
            s += wi[k2 * G4H + tid] * hs[l - 1][k2];
        }
        const float* wh = wThh + (long)l * (G4H * HH);
#pragma unroll 4
        for (int k2 = 0; k2 < HH; k2++)
          s += wh[k2 * G4H + tid] * hs[l][k2];
        gates[tid] = s;
      }
      __syncthreads();
      if (tid < HH) {
        const float ig = 1.f / (1.f + expf(-gates[tid]));
        const float fg = 1.f / (1.f + expf(-gates[HH + tid]));
        const float gg = tanhf(gates[2 * HH + tid]);
        const float og = 1.f / (1.f + expf(-gates[3 * HH + tid]));
        const float cn = fg * cs[l][tid] + ig * gg;
        cs[l][tid] = cn;
        hs[l][tid] = og * tanhf(cn);
      }
      __syncthreads();
    }
    if (tid < HH) out[(long)t * HH + tid] = hs[LL - 1][tid];
  }
  for (int i = tid; i < LL * HH; i += 512) {
    const int l = i / HH, k2 = i % HH;
    hc[i] = hs[l][k2];
    hc[LL * HH + i] = cs[l][k2];
  }
}

// ---------------------------------------------------------------------------
// RGAT attention projections: xq/xk[n,r,h] = sum_f xW[n,r,f] * {q,k}[f,h]
// one block per node, one wave per relation, cross-lane shfl reduce.
// ---------------------------------------------------------------------------
__global__ __launch_bounds__(256) void attn_proj(
    const float* __restrict__ xW, const float* __restrict__ q,
    const float* __restrict__ k, float* __restrict__ xq,
    float* __restrict__ xk)
{
  const int n    = blockIdx.x;
  const int r    = threadIdx.x >> 5;
  const int lane = threadIdx.x & 31;
  const float* xw = xW + (long)n * (RR * HC) + r * HC;

  float aq[HEADS] = {}, ak[HEADS] = {};
  for (int f = lane; f < HC; f += 32) {
    const float v = xw[f];
    const float* qf = q + f * HEADS;
    const float* kf = k + f * HEADS;
#pragma unroll
    for (int h = 0; h < HEADS; h++) { aq[h] += v * qf[h]; ak[h] += v * kf[h]; }
  }
#pragma unroll
  for (int off = 16; off; off >>= 1) {
#pragma unroll
    for (int h = 0; h < HEADS; h++) {
      aq[h] += __shfl_xor(aq[h], off, 32);
      ak[h] += __shfl_xor(ak[h], off, 32);
    }
  }
  if (lane == 0) {
    float* pq = xq + ((long)n * RR + r) * HEADS;
    float* pk = xk + ((long)n * RR + r) * HEADS;
#pragma unroll
    for (int h = 0; h < HEADS; h++) { pq[h] = aq[h]; pk[h] = ak[h]; }
  }
}

// order-preserving float<->uint encoding for atomic max
__device__ __forceinline__ unsigned enc_ord(float v) {
  unsigned u = __float_as_uint(v);
  return (u & 0x80000000u) ? ~u : (u | 0x80000000u);
}
__device__ __forceinline__ float dec_ord(unsigned u) {
  unsigned b = (u & 0x80000000u) ? (u & 0x7FFFFFFFu) : ~u;
  return __uint_as_float(b);
}

__global__ void init_attn(unsigned* __restrict__ amax, float* __restrict__ den,
                          int n)
{
  const int i = blockIdx.x * blockDim.x + threadIdx.x;
  if (i < n) { amax[i] = 0u; den[i] = 0.f; }   // 0 is below enc(-inf) == 0x007FFFFF
}

__global__ void fzero(float* __restrict__ p, int n)
{
  const int i = blockIdx.x * blockDim.x + threadIdx.x;
  if (i < n) p[i] = 0.f;
}

__global__ void edge_logits(const int* __restrict__ src,
                            const int* __restrict__ dst,
                            const int* __restrict__ et,
                            const float* __restrict__ xq,
                            const float* __restrict__ xk,
                            float* __restrict__ lg,
                            unsigned* __restrict__ amax, int E)
{
  const int idx = blockIdx.x * blockDim.x + threadIdx.x;
  if (idx >= E * HEADS) return;
  const int e = idx >> 3, h = idx & 7;
  const int s = src[e], d = dst[e], t = et[e];
  float v = xq[((long)d * RR + t) * HEADS + h] +
            xk[((long)s * RR + t) * HEADS + h];
  v = (v > 0.f) ? v : 0.2f * v;                  // leaky_relu 0.2
  lg[idx] = v;
  atomicMax(&amax[d * HEADS + h], enc_ord(v));
}

__global__ void edge_expsum(const int* __restrict__ dst,
                            float* __restrict__ lg,
                            const unsigned* __restrict__ amax,
                            float* __restrict__ den, int E)
{
  const int idx = blockIdx.x * blockDim.x + threadIdx.x;
  if (idx >= E * HEADS) return;
  const int e = idx >> 3, h = idx & 7;
  const int d = dst[e];
  const float ex = expf(lg[idx] - dec_ord(amax[d * HEADS + h]));
  lg[idx] = ex;
  atomicAdd(&den[d * HEADS + h], ex);
}

// one block per edge: acc[dst,c] += xW[src,et,c] * alpha[e, c/100]
__global__ __launch_bounds__(256) void edge_scatter(
    const int* __restrict__ src, const int* __restrict__ dst,
    const int* __restrict__ et, const float* __restrict__ xW,
    const float* __restrict__ lg, const float* __restrict__ den,
    float* __restrict__ acc)
{
  const int e = blockIdx.x;
  const int s = src[e], d = dst[e], t = et[e];
  const float* xws = xW + ((long)s * RR + t) * HC;
  float* accd = acc + (long)d * HC;
  for (int c = threadIdx.x; c < HC; c += blockDim.x) {
    const int h = c / HH;
    const float alpha = lg[e * HEADS + h] / (den[d * HEADS + h] + 1e-16f);
    atomicAdd(&accd[c], xws[c] * alpha);
  }
}

__global__ void bias_relu(float* __restrict__ p, const float* __restrict__ b,
                          int n)
{
  const int i = blockIdx.x * blockDim.x + threadIdx.x;
  if (i < n) p[i] = fmaxf(p[i] + b[i % HC], 0.f);
}

__global__ void final_readout(const float* __restrict__ lout,
                              const int* __restrict__ qidx,
                              const float* __restrict__ lin_w,
                              const float* __restrict__ lin_b,
                              float* __restrict__ out)
{
  __shared__ float mean[HH];
  const int tid = threadIdx.x;
  if (tid < HH) {
    float s = 0.f;
    for (int j = 0; j < 64; j++) s += lout[(long)qidx[j] * HH + tid];
    mean[tid] = s * (1.0f / 64.0f);
  }
  __syncthreads();
  if (tid < 3) {
    float s = lin_b[tid];
    for (int k2 = 0; k2 < HH; k2++) s += lin_w[tid * HH + k2] * mean[k2];
    out[tid] = s;
  }
}

// ---------------------------------------------------------------------------
extern "C" void kernel_launch(void* const* d_in, const int* in_sizes, int n_in,
                              void* d_out, int out_size, void* d_ws,
                              size_t ws_size, hipStream_t stream)
{
  (void)in_sizes; (void)n_in; (void)out_size; (void)ws_size;

  const int*   x       = (const int*)d_in[0];
  const int*   ei      = (const int*)d_in[1];   // [2,E]: src then dst
  const int*   etype   = (const int*)d_in[2];
  const int*   qidx    = (const int*)d_in[3];
  const float* emb     = (const float*)d_in[4];
  const float* init_w  = (const float*)d_in[5];
  const float* init_b  = (const float*)d_in[6];
  const float* gat_W   = (const float*)d_in[7];
  const float* gat_q   = (const float*)d_in[8];
  const float* gat_k   = (const float*)d_in[9];
  const float* gat_b   = (const float*)d_in[10];
  const float* w_ih0   = (const float*)d_in[11];
  const float* w_ihr   = (const float*)d_in[12];
  const float* w_hh    = (const float*)d_in[13];
  const float* b_ih    = (const float*)d_in[14];
  const float* b_hh    = (const float*)d_in[15];
  const float* lin_w   = (const float*)d_in[16];
  const float* lin_b   = (const float*)d_in[17];
  float* out = (float*)d_out;

  // workspace layout (floats)
  float* ws    = (float*)d_ws;
  float* seq   = ws;                        // 4096*800
  float* xproj = seq   + (size_t)NN * HC;   // 4096*400
  float* lout  = xproj + (size_t)NN * G4H;  // 4096*100
  float* acc   = lout  + (size_t)NN * HH;   // 4096*800 (rgat out / next input)
  float* xW    = acc   + (size_t)NN * HC;   // 4096*8*800
  float* xq    = xW    + (size_t)NN * RR * HC;      // 4096*8*8
  float* xk    = xq    + (size_t)NN * RR * HEADS;   // 4096*8*8
  float* lg    = xk    + (size_t)NN * RR * HEADS;   // 65536*8
  float* amaxf = lg    + (size_t)EE * HEADS;        // 4096*8 (as uint)
  float* den   = amaxf + (size_t)NN * HEADS;        // 4096*8
  float* wTih  = den   + (size_t)NN * HEADS;        // 7*100*400
  float* wThh  = wTih  + (size_t)7 * HH * G4H;      // 8*100*400
  float* hc    = wThh  + (size_t)LL * HH * G4H;     // 2*8*100
  unsigned* amax = (unsigned*)amaxf;

  const int* esrc = ei;
  const int* edst = ei + EE;

  // LSTM weight transposes (coalesced matvecs inside the scan)
  transpose_lstm<<<(7 * G4H * HH + 255) / 256, 256, 0, stream>>>(w_ihr, wTih, 7);
  transpose_lstm<<<(LL * G4H * HH + 255) / 256, 256, 0, stream>>>(w_hh, wThh, LL);

  // seq = relu(emb[x] @ init_w^T + init_b)        [4096,800], K=300
  wmma_gemm_nt<<<dim3(13, 64, 1), 256, 0, stream>>>(
      emb, EMB, x, init_w, EMB, 1, 0, seq, HC, 0, init_b, NN, HC, EMB, 1);

  // xproj = seq @ w_ih0^T                          [4096,400], K=800
  wmma_gemm_nt<<<dim3(7, 64, 1), 256, 0, stream>>>(
      seq, HC, nullptr, w_ih0, HC, 1, 0, xproj, G4H, 0, nullptr, NN, G4H, HC, 0);

  lstm_scan<<<1, 512, 0, stream>>>(xproj, wTih, wThh, b_ih, b_hh, lout, hc, NN, 1);

  for (int l = 0; l < 3; l++) {
    // xW[:,r,:] = lout @ W[l,r]  (W[l,r] is [d=100, f=800] -> sN=1, sK=800)
    wmma_gemm_nt<<<dim3(13, 64, RR), 256, 0, stream>>>(
        lout, HH, nullptr, gat_W + (long)l * RR * HH * HC, 1, HC,
        (long)HH * HC, xW, RR * HC, HC, nullptr, NN, HC, HH, 0);

    attn_proj<<<NN, 256, 0, stream>>>(xW, gat_q + (long)l * HC * HEADS,
                                      gat_k + (long)l * HC * HEADS, xq, xk);

    init_attn<<<(NN * HEADS + 255) / 256, 256, 0, stream>>>(amax, den, NN * HEADS);
    fzero<<<(NN * HC + 255) / 256, 256, 0, stream>>>(acc, NN * HC);

    edge_logits<<<(EE * HEADS + 255) / 256, 256, 0, stream>>>(
        esrc, edst, etype, xq, xk, lg, amax, EE);
    edge_expsum<<<(EE * HEADS + 255) / 256, 256, 0, stream>>>(
        edst, lg, amax, den, EE);
    edge_scatter<<<EE, 256, 0, stream>>>(esrc, edst, etype, xW, lg, den, acc);

    bias_relu<<<(NN * HC + 255) / 256, 256, 0, stream>>>(
        acc, gat_b + (long)l * HC, NN * HC);

    // next LSTM phase: xproj = g @ w_ih0^T, then scan continuing (h,c)
    wmma_gemm_nt<<<dim3(7, 64, 1), 256, 0, stream>>>(
        acc, HC, nullptr, w_ih0, HC, 1, 0, xproj, G4H, 0, nullptr, NN, G4H, HC, 0);
    lstm_scan<<<1, 512, 0, stream>>>(xproj, wTih, wThh, b_ih, b_hh, lout, hc, NN, 0);
  }

  final_readout<<<1, 128, 0, stream>>>(lout, qidx, lin_w, lin_b, out);
}
